// DCNv2_69810398429897
// MI455X (gfx1250) — matrix operand order repaired and checked by reading, not compiled, not run on hardware
//
#include <hip/hip_runtime.h>
#include <hip/hip_bf16.h>
#include <math.h>

// ---------------- problem constants ----------------
#define BB 4
#define CC 256
#define HH 64
#define WW 64
#define OO 256
#define KKT 9            // 3x3 taps
#define KDIM (CC*KKT)    // 2304, k = p*256 + c  (tap-major)
#define MTOT (BB*HH*WW)  // 16384 pixels
#define HWS  (HH*WW)     // 4096
#define NCHUNK 72        // 2304 / 32 K-chunks

// ---------------- vector types ----------------
typedef __attribute__((ext_vector_type(16))) __bf16 v16bf;
typedef __attribute__((ext_vector_type(8)))  __bf16 v8bf;
typedef __attribute__((ext_vector_type(4)))  __bf16 v4bf;
typedef __attribute__((ext_vector_type(8)))  float  v8f;
typedef __attribute__((ext_vector_type(4)))  float  v4f;
typedef __attribute__((ext_vector_type(4)))  unsigned int v4u;
typedef __attribute__((ext_vector_type(8)))  int    v8i;
typedef __attribute__((ext_vector_type(4)))  int    v4i;

union AU { v16bf v; v8bf h[2]; };

#if __has_builtin(__builtin_amdgcn_tensor_load_to_lds)
#define HAVE_TDM 1
#else
#define HAVE_TDM 0
#endif

// fp32 -> bf16 (round-to-nearest-even) via integer ops (no libcall risk)
static __device__ inline __bf16 f2bf(float f) {
    unsigned u = __builtin_bit_cast(unsigned, f);
    unsigned r = (u + 0x7FFFu + ((u >> 16) & 1u)) >> 16;
    unsigned short s = (unsigned short)r;
    return __builtin_bit_cast(__bf16, s);
}

static __device__ inline float fast_sigmoid(float v) {
    float e = __expf(-v);
#if __has_builtin(__builtin_amdgcn_rcpf)
    return __builtin_amdgcn_rcpf(1.f + e);
#else
    return 1.f / (1.f + e);
#endif
}

// ---------------- workspace layout (bytes) ----------------
// xt : NHWC fp32            16,777,216
// OM : [16384][32] fp32      2,097,152   (off y/x ch 0..17, sigmoid(mask) ch 18..26)
// B1 : [2304][32]  bf16        147,456
// B2 : [2304][256] bf16      1,179,648
#define WS_XT 0
#define WS_OM 16777216
#define WS_B1 18874368
#define WS_B2 19021824

// =====================================================================
// Kernel 0: NCHW -> NHWC transpose of x (LDS tiled, coalesced both sides)
// =====================================================================
__global__ __launch_bounds__(256) void k_transpose(const float* __restrict__ x,
                                                   float* __restrict__ xt) {
    __shared__ float tile[32][33];
    int b  = blockIdx.z;
    int s0 = blockIdx.x * 32;
    int c0 = blockIdx.y * 32;
    #pragma unroll
    for (int i = 0; i < 4; ++i) {
        int c = c0 + threadIdx.y + i * 8;
        tile[threadIdx.y + i * 8][threadIdx.x] =
            x[(size_t)(b * CC + c) * HWS + s0 + threadIdx.x];
    }
    __syncthreads();
    #pragma unroll
    for (int i = 0; i < 4; ++i) {
        int s = s0 + threadIdx.y + i * 8;
        xt[(size_t)(b * HWS + s) * CC + c0 + threadIdx.x] =
            tile[threadIdx.x][threadIdx.y + i * 8];
    }
}

// =====================================================================
// Kernel 1: pack weights to bf16, [k = p*256+c][n] layout
// =====================================================================
__global__ __launch_bounds__(256) void k_pack(const float* __restrict__ w_off,
                                              const float* __restrict__ w_mask,
                                              const float* __restrict__ w_dcn,
                                              unsigned short* __restrict__ B1,
                                              unsigned short* __restrict__ B2) {
    int i = blockIdx.x * 256 + threadIdx.x;
    const int N2 = KDIM * OO;
    if (i < N2) {
        int k = i >> 8, n = i & 255;
        int p = k >> 8, c = k & 255;          // k = p*256 + c
        float v = w_dcn[((size_t)n * CC + c) * KKT + p];
        B2[i] = __builtin_bit_cast(unsigned short, f2bf(v));
    } else {
        int i2 = i - N2;                      // i2 < 2304*32
        int k = i2 >> 5, n = i2 & 31;
        int p = k >> 8, c = k & 255;
        float v = 0.f;
        if (n < 18)      v = w_off[((size_t)n * CC + c) * KKT + p];
        else if (n < 27) v = w_mask[((size_t)(n - 18) * CC + c) * KKT + p];
        B1[i2] = __builtin_bit_cast(unsigned short, f2bf(v));
    }
}

// =====================================================================
// Kernel 2: offset/mask conv as WMMA GEMM (M=16384, N=32(pad 27), K=2304)
// =====================================================================
#define AP1 40   // A tile pitch (shorts): 80B -> conflict-free strided ds_load_b128
__global__ __launch_bounds__(256) void k_offmask(const float* __restrict__ Xt,
                                                 const __bf16* __restrict__ B1,
                                                 const float* __restrict__ b_off,
                                                 const float* __restrict__ b_mask,
                                                 float* __restrict__ OM) {
    __shared__ __align__(16) __bf16 As[64 * AP1];
    __shared__ int   sIdx[64];
    __shared__ float sW[64];

    int t = threadIdx.x, lane = t & 31, wave = t >> 5;
    int wm = wave & 3, wn = wave >> 2;
    int Mbase = blockIdx.x * 64;
    v8f acc = {0.f, 0.f, 0.f, 0.f, 0.f, 0.f, 0.f, 0.f};

    for (int p = 0; p < KKT; ++p) {
        __syncthreads();
        if (t < 64) {                       // per-pixel tap params (zero-pad conv)
            int m = Mbase + t;
            int b = m >> 12, s = m & 4095, h = s >> 6, w = s & 63;
            int iy = h + p / 3 - 1, ix = w + p % 3 - 1;
            bool valid = (iy >= 0) & (iy < HH) & (ix >= 0) & (ix < WW);
            int cy = min(max(iy, 0), HH - 1), cx = min(max(ix, 0), WW - 1);
            sIdx[t] = ((b * HH + cy) * WW + cx) * CC;
            sW[t]   = valid ? 1.f : 0.f;
        }
        __syncthreads();
        for (int cb = 0; cb < 8; ++cb) {
            int cbase = cb * 32;
            // ---- produce im2col A tile (64 x 32 bf16) ----
            {
                int c4 = (t & 7) * 4;
                int r0 = (t >> 3) * 2;
                #pragma unroll
                for (int e = 0; e < 2; ++e) {
                    int r = r0 + e;
                    v4f g = *(const v4f*)(Xt + sIdx[r] + cbase + c4);
                    float wgt = sW[r];
                    v4bf ob;
                    ob[0] = f2bf(g[0] * wgt); ob[1] = f2bf(g[1] * wgt);
                    ob[2] = f2bf(g[2] * wgt); ob[3] = f2bf(g[3] * wgt);
                    *(v4bf*)&As[r * AP1 + c4] = ob;
                }
            }
            __syncthreads();
            // ---- consume: one v_wmma_f32_16x16x32_bf16 per wave ----
            int kglob = p * 256 + cbase;
            int row = wm * 16 + (lane & 15);
            int kA  = (lane < 16) ? 0 : 8;
            AU a;
            a.h[0] = *(const v8bf*)&As[row * AP1 + kA];
            a.h[1] = *(const v8bf*)&As[row * AP1 + kA + 16];
            AU bfr;
            const __bf16* bp = B1 + (size_t)(kglob + ((lane >> 4) << 3)) * 32 + wn * 16 + (lane & 15);
            #pragma unroll
            for (int s = 0; s < 8; ++s) {
                bfr.v[s]     = bp[s * 32];
                bfr.v[8 + s] = bp[(16 + s) * 32];
            }
            acc = __builtin_amdgcn_wmma_f32_16x16x32_bf16(
                false, a.v, false, bfr.v, (short)0, acc, false, false);
            __syncthreads();
        }
    }
    // epilogue: bias + sigmoid(mask channels), store [pixel][32]
    int n = wn * 16 + (lane & 15);
    float bias = (n < 18) ? b_off[n] : ((n < 27) ? b_mask[n - 18] : 0.f);
    bool sig = (n >= 18) & (n < 27);
    int m0 = Mbase + wm * 16 + ((lane >> 4) << 3);
    #pragma unroll
    for (int r = 0; r < 8; ++r) {
        float v = acc[r] + bias;
        if (sig) v = fast_sigmoid(v);
        OM[(size_t)(m0 + r) * 32 + n] = v;
    }
}

// =====================================================================
// Kernel 3: main deformable conv as implicit WMMA GEMM
// M=16384, N=256, K=2304; 256 blocks, BM=64, BN=256
// 8 waves = 4 M-tiles(16) x 2 N-halves(128); 8 wmma / wave / K-chunk
// B tiles: TDM double-buffered global->LDS with HW padding (pitch 264 shorts)
// =====================================================================
#define AP2 40    // A pitch (shorts)
#define BP2 264   // B pitch (shorts): 528B = TDM pad(4 DW per 128 DW row)
__global__ __launch_bounds__(256) void k_dcn(const float* __restrict__ Xt,
                                             const float* __restrict__ OM,
                                             const __bf16* __restrict__ B2,
                                             const float* __restrict__ b_dcn,
                                             float* __restrict__ out) {
    __shared__ __align__(16) __bf16 As[64 * AP2];
    __shared__ __align__(16) __bf16 Bs[2][32 * BP2];
    __shared__ int   sIdx[64][4];
    __shared__ float sW[64][4];

    int t = threadIdx.x, lane = t & 31, wave = t >> 5;
    int wm = wave & 3, wn = wave >> 2;
    int Mbase = blockIdx.x * 64;

    v8f acc[8];
    #pragma unroll
    for (int j = 0; j < 8; ++j) acc[j] = (v8f){0.f,0.f,0.f,0.f,0.f,0.f,0.f,0.f};

#if HAVE_TDM
    const bool tdmWave = (wave == 0);
    // ---- TDM descriptor: 2-D tile 32 rows x 256 bf16, row stride 512B,
    //      LDS pad 4 DWORDs every 128 DWORDs -> LDS pitch 528B (= BP2)
    auto issueB = [&](int kc, int buf) {
        unsigned long long ga =
            (unsigned long long)(uintptr_t)B2 + (unsigned long long)kc * 16384ull;
        unsigned lds = (unsigned)(uintptr_t)&Bs[buf][0];
        v4u g0 = { 1u,                                   // count=1 (valid user D#)
                   lds,                                  // lds_addr
                   (unsigned)ga,                         // global_addr[31:0]
                   (unsigned)((ga >> 32) & 0x01FFFFFFu)  // global_addr[56:32]
                       | 0x80000000u };                  // type=2 ("image")
        v8i g1 = { (int)((1u << 16)     // data_size = 2B
                       | (1u << 20)     // pad_enable
                       | (6u << 22)     // pad_interval: 128 DWORDs (one row)
                       | (3u << 25)),   // pad_amount:   4 DWORDs (16B)
                   (int)(256u << 16),   // tensor_dim0 = 256 (lo16 in [63:48])
                   (int)(32u  << 16),   // tensor_dim1 = 32  (lo16 in [95:80])
                   (int)(256u << 16),   // tile_dim0   = 256 ([127:112])
                   32,                  // tile_dim1 = 32, tile_dim2 = 0
                   256,                 // tensor_dim0_stride = 256 elements
                   0, 0 };
        v4i z4 = {0, 0, 0, 0};
        v8i z8 = {0, 0, 0, 0, 0, 0, 0, 0};
        __builtin_amdgcn_tensor_load_to_lds(g0, g1, z4, z4, z8, 0);
    };
    if (tdmWave) issueB(0, 0);
#endif

    for (int p = 0; p < KKT; ++p) {
        if (t < 64) {   // bilinear params: 4 gather bases + 4 mask-scaled weights
            int m = Mbase + t;
            int b = m >> 12, s = m & 4095, h = s >> 6, w = s & 63;
            const float* om = OM + (size_t)m * 32;
            float oy = om[2 * p], ox = om[2 * p + 1], mk = om[18 + p];
            float py = (float)(h + p / 3 - 1) + oy;
            float px = (float)(w + p % 3 - 1) + ox;
            float y0f = floorf(py), x0f = floorf(px);
            int y0 = (int)y0f, x0 = (int)x0f;
            float wy = py - y0f, wx = px - x0f;
            float wgt[4] = { (1.f - wy) * (1.f - wx) * mk, (1.f - wy) * wx * mk,
                             wy * (1.f - wx) * mk,         wy * wx * mk };
            int dy[4] = {0, 0, 1, 1}, dx[4] = {0, 1, 0, 1};
            #pragma unroll
            for (int i = 0; i < 4; ++i) {
                int iy = y0 + dy[i], ix = x0 + dx[i];
                bool valid = (iy >= 0) & (iy < HH) & (ix >= 0) & (ix < WW);
                int cy = min(max(iy, 0), HH - 1), cx = min(max(ix, 0), WW - 1);
                int idx = ((b * HH + cy) * WW + cx) * CC;
                sIdx[t][i] = idx;
                sW[t][i]   = valid ? wgt[i] : 0.f;
                __builtin_prefetch(Xt + idx, 0, 1);   // row reused by 8 chunks
            }
        }
        __syncthreads();
        for (int cb = 0; cb < 8; ++cb) {
            int kc = p * 8 + cb;
            int cur = kc & 1;
            int cbase = cb * 32;
            // ---- produce A: sampled*mask, 64 pixels x 32 channels, bf16 ----
            {
                int c4 = (t & 7) * 4;
                int r0 = (t >> 3) * 2;
                #pragma unroll
                for (int e = 0; e < 2; ++e) {
                    int r = r0 + e;
                    v4f g = {0.f, 0.f, 0.f, 0.f};
                    #pragma unroll
                    for (int i = 0; i < 4; ++i) {
                        v4f xv = *(const v4f*)(Xt + sIdx[r][i] + cbase + c4);
                        g += sW[r][i] * xv;
                    }
                    v4bf ob;
                    ob[0] = f2bf(g[0]); ob[1] = f2bf(g[1]);
                    ob[2] = f2bf(g[2]); ob[3] = f2bf(g[3]);
                    *(v4bf*)&As[r * AP2 + c4] = ob;
                }
            }
#if HAVE_TDM
            // ---- async: kick B tile for chunk kc+1, wait for chunk kc ----
            if (tdmWave) {
                if (kc < NCHUNK - 1) {
                    issueB(kc + 1, cur ^ 1);
                    __builtin_amdgcn_s_wait_tensorcnt(1);
                } else {
                    __builtin_amdgcn_s_wait_tensorcnt(0);
                }
            }
#else
            // ---- fallback: cooperative global->LDS copy of B tile ----
            {
                int kglob = kc * 32;
                int row = t >> 3;
                int col = (t & 7) * 32;   // shorts
                const uint4* src = (const uint4*)((const unsigned short*)B2 +
                                                  (size_t)(kglob + row) * OO + col);
                #pragma unroll
                for (int q = 0; q < 4; ++q)
                    *(uint4*)&Bs[cur][row * BP2 + col + q * 8] = src[q];
            }
#endif
            __syncthreads();
            // ---- consume: 8 x v_wmma_f32_16x16x32_bf16 per wave ----
            AU a;
            {
                int row = wm * 16 + (lane & 15);
                int kA  = (lane < 16) ? 0 : 8;
                a.h[0] = *(const v8bf*)&As[row * AP2 + kA];
                a.h[1] = *(const v8bf*)&As[row * AP2 + kA + 16];
            }
            #pragma unroll
            for (int j = 0; j < 8; ++j) {
                AU bfr;
                int ncol = wn * 128 + j * 16;
                int kB = (lane >> 4) << 3;
                #pragma unroll
                for (int s = 0; s < 8; ++s) {
                    bfr.v[s]     = Bs[cur][(kB + s) * BP2 + ncol + (lane & 15)];
                    bfr.v[8 + s] = Bs[cur][(kB + 16 + s) * BP2 + ncol + (lane & 15)];
                }
                acc[j] = __builtin_amdgcn_wmma_f32_16x16x32_bf16(
                    false, a.v, false, bfr.v, (short)0, acc[j], false, false);
            }
            __syncthreads();
        }
    }
    // epilogue: + b_dcn, scatter to NCHW output (contiguous 32B per lane per tile)
    int m0 = Mbase + wm * 16 + ((lane >> 4) << 3);
    int b  = m0 >> 12, sp = m0 & 4095;
    #pragma unroll
    for (int j = 0; j < 8; ++j) {
        int n = wn * 128 + j * 16 + (lane & 15);
        float bias = b_dcn[n];
        float* op = out + ((size_t)(b * OO + n)) * HWS + sp;
        v4f lo = { acc[j][0] + bias, acc[j][1] + bias, acc[j][2] + bias, acc[j][3] + bias };
        v4f hi = { acc[j][4] + bias, acc[j][5] + bias, acc[j][6] + bias, acc[j][7] + bias };
        *(v4f*)op       = lo;
        *(v4f*)(op + 4) = hi;
    }
}

// =====================================================================
extern "C" void kernel_launch(void* const* d_in, const int* in_sizes, int n_in,
                              void* d_out, int out_size, void* d_ws, size_t ws_size,
                              hipStream_t stream) {
    const float* x      = (const float*)d_in[0];
    const float* w_off  = (const float*)d_in[1];
    const float* b_off  = (const float*)d_in[2];
    const float* w_mask = (const float*)d_in[3];
    const float* b_mask = (const float*)d_in[4];
    const float* w_dcn  = (const float*)d_in[5];
    const float* b_dcn  = (const float*)d_in[6];
    float* out = (float*)d_out;

    char* ws = (char*)d_ws;                       // needs ~20.2 MB scratch
    float*          xt = (float*)(ws + WS_XT);
    float*          om = (float*)(ws + WS_OM);
    unsigned short* B1 = (unsigned short*)(ws + WS_B1);
    unsigned short* B2 = (unsigned short*)(ws + WS_B2);

    k_transpose<<<dim3(HWS / 32, CC / 32, BB), dim3(32, 8), 0, stream>>>(x, xt);
    k_pack<<<dim3((KDIM * OO + KDIM * 32) / 256), dim3(256), 0, stream>>>(
        w_off, w_mask, w_dcn, B1, B2);
    k_offmask<<<dim3(MTOT / 64), dim3(256), 0, stream>>>(
        xt, (const __bf16*)B1, b_off, b_mask, om);
    k_dcn<<<dim3(MTOT / 64), dim3(256), 0, stream>>>(
        xt, om, (const __bf16*)B2, b_dcn, out);
}
